// AttnRNN_75823352644232
// MI455X (gfx1250) — matrix operand order repaired and checked by reading, not compiled, not run on hardware
//
#include <hip/hip_runtime.h>
#include <hip/hip_bf16.h>
#include <math.h>

// ---------------------------------------------------------------------------
// Types / helpers
// ---------------------------------------------------------------------------
typedef __bf16 bf16_t;
typedef __attribute__((ext_vector_type(8)))  bf16_t bfx8;
typedef __attribute__((ext_vector_type(16))) bf16_t bfx16;
typedef __attribute__((ext_vector_type(8)))  float  fx8;

#define DEV __device__ __forceinline__

DEV bf16_t f2bf(float f) {
  union { float f; unsigned u; } a; a.f = f;
  unsigned u = a.u;
  unsigned r = u + 0x7FFFu + ((u >> 16) & 1u);   // round-to-nearest-even
  unsigned short h = (unsigned short)(r >> 16);
  bf16_t o; __builtin_memcpy(&o, &h, 2); return o;
}
DEV float bf2f(bf16_t b) {
  unsigned short h; __builtin_memcpy(&h, &b, 2);
  union { unsigned u; float f; } a; a.u = ((unsigned)h) << 16; return a.f;
}
DEV float sigmoidf_(float x) { return 1.0f / (1.0f + expf(-x)); }

// A-fragment (16x32 bf16, M x K). Lane l (0-15): row m0+l, K {k0..k0+7, k0+16..k0+23}
// Lane l (16-31): row m0+(l&15), K {k0+8..k0+15, k0+24..k0+31}
DEV bfx16 frag_a(const bf16_t* __restrict__ A, long m0, long lda, long k0, int rowsMax) {
  const int lane = threadIdx.x & 31;
  long r = m0 + (lane & 15);
  if (r >= rowsMax) r = rowsMax - 1;
  const bf16_t* p = A + r * lda + k0 + (long)((lane >> 4) << 3);
  bfx8 lo = *(const bfx8*)(p);
  bfx8 hi = *(const bfx8*)(p + 16);
  bfx16 v;
#pragma unroll
  for (int i = 0; i < 8; ++i) { v[i] = lo[i]; v[i + 8] = hi[i]; }
  return v;
}

// B-fragment (32x16 bf16, K x N) built from W (Nout x K, row-major) used as B = W^T.
// Lane l (0-15): col n0+l, K {k0..k0+7, k0+16..k0+23} ; lane (16-31): col n0+(l&15), K {k0+8..k0+15, k0+24..k0+31}
DEV bfx16 frag_b(const bf16_t* __restrict__ W, long n0, long ldw, long k0, int colsMax) {
  const int lane = threadIdx.x & 31;
  long r = n0 + (lane & 15);
  if (r >= colsMax) r = colsMax - 1;
  const bf16_t* p = W + r * ldw + k0 + (long)((lane >> 4) << 3);
  bfx8 lo = *(const bfx8*)(p);
  bfx8 hi = *(const bfx8*)(p + 16);
  bfx16 v;
#pragma unroll
  for (int i = 0; i < 8; ++i) { v[i] = lo[i]; v[i + 8] = hi[i]; }
  return v;
}

DEV fx8 wmma_bf16(bfx16 a, bfx16 b, fx8 c) {
  return __builtin_amdgcn_wmma_f32_16x16x32_bf16(false, a, false, b, (short)0, c, false, false);
}

// ---------------------------------------------------------------------------
// f32 -> bf16 weight conversion (optional transpose, optional row stride)
// ---------------------------------------------------------------------------
__global__ void conv_bf16_kernel(const float* __restrict__ src, bf16_t* __restrict__ dst,
                                 long rows, long cols, long srcStride, int transpose) {
  long idx = (long)blockIdx.x * 256 + threadIdx.x;
  long tot = rows * cols;
  if (idx >= tot) return;
  long r = idx / cols, c = idx % cols;
  float v = src[r * srcStride + c];
  if (transpose) dst[c * rows + r] = f2bf(v);
  else           dst[r * cols + c] = f2bf(v);
}

// qbias[j] = attn_out_b[j] + sum_k attn_out_W[j, 1024+k] * q[k]   (constant query half)
__global__ void qbias_kernel(const float* __restrict__ W, const float* __restrict__ b,
                             const float* __restrict__ q, float* __restrict__ out) {
  int j = blockIdx.x * blockDim.x + threadIdx.x;
  if (j < 1024) {
    float s = b[j];
    const float* row = W + (long)j * 2048 + 1024;
    for (int k = 0; k < 1024; ++k) s += row[k] * q[k];
    out[j] = s;
  }
}

// ---------------------------------------------------------------------------
// Embedding gather -> bf16 rows
// ---------------------------------------------------------------------------
__global__ void embed_gather_kernel(const float* __restrict__ embed, const int* __restrict__ x,
                                    bf16_t* __restrict__ xe, long nTok) {
  long row = blockIdx.x;
  if (row >= nTok) return;
  int tok = x[row];
  const float* src = embed + (long)tok * 256;
  bf16_t* dst = xe + row * 256;
  for (int i = threadIdx.x; i < 256; i += 64) dst[i] = f2bf(src[i]);
}

// ---------------------------------------------------------------------------
// Generic GEMM: C(MxNt) = A(MxK,bf16) * W(NtxK,bf16)^T [+bias][+bias2][tanh]
// Block = 4 waves; each wave: 32 rows x 64 cols -> 8 WMMAs per k-step.
// All epilogue behavior resolved at compile time (no per-element branches).
// ---------------------------------------------------------------------------
template <bool HB1, bool HB2, bool TANH, bool OUTF, bool OUTB>
__global__ __launch_bounds__(128)
void gemm_bf16_kernel(const bf16_t* __restrict__ A, long lda,
                      const bf16_t* __restrict__ W, long ldw,
                      const float* __restrict__ bias, const float* __restrict__ bias2,
                      float* __restrict__ Cf, bf16_t* __restrict__ Cb, long ldc,
                      int M, int K, int Nt) {
  const int lane = threadIdx.x & 31;
  const int wv   = threadIdx.x >> 5;
  const long n0 = (long)blockIdx.x * 64;
  const long m0 = (long)blockIdx.y * 128 + (long)wv * 32;

  fx8 acc[2][4];
#pragma unroll
  for (int i = 0; i < 2; ++i)
#pragma unroll
    for (int j = 0; j < 4; ++j) acc[i][j] = (fx8)0.0f;

  for (long k0 = 0; k0 < K; k0 += 32) {
    bfx16 a0 = frag_a(A, m0,      lda, k0, M);
    bfx16 a1 = frag_a(A, m0 + 16, lda, k0, M);
    bfx16 b0 = frag_b(W, n0,      ldw, k0, Nt);
    bfx16 b1 = frag_b(W, n0 + 16, ldw, k0, Nt);
    bfx16 b2 = frag_b(W, n0 + 32, ldw, k0, Nt);
    bfx16 b3 = frag_b(W, n0 + 48, ldw, k0, Nt);
    acc[0][0] = wmma_bf16(a0, b0, acc[0][0]);
    acc[0][1] = wmma_bf16(a0, b1, acc[0][1]);
    acc[0][2] = wmma_bf16(a0, b2, acc[0][2]);
    acc[0][3] = wmma_bf16(a0, b3, acc[0][3]);
    acc[1][0] = wmma_bf16(a1, b0, acc[1][0]);
    acc[1][1] = wmma_bf16(a1, b1, acc[1][1]);
    acc[1][2] = wmma_bf16(a1, b2, acc[1][2]);
    acc[1][3] = wmma_bf16(a1, b3, acc[1][3]);
  }

  const int hi = lane >> 4, nn = lane & 15;
  // Hoisted per-column bias adds + validity (uniform across the v/row loop).
  float badd[4];
  bool  colok[4];
#pragma unroll
  for (int j = 0; j < 4; ++j) {
    long col = n0 + j * 16 + nn;
    colok[j] = (col < Nt);
    long cs = colok[j] ? col : (long)(Nt - 1);
    float b = 0.f;
    if (HB1) b += bias[cs];
    if (HB2) b += bias2[cs];
    badd[j] = b;
  }
#pragma unroll
  for (int i = 0; i < 2; ++i)
#pragma unroll
    for (int v = 0; v < 8; ++v) {
      long row = m0 + i * 16 + v + hi * 8;
      if (row < M) {
#pragma unroll
        for (int j = 0; j < 4; ++j) {
          if (colok[j]) {
            long col = n0 + j * 16 + nn;
            float x = acc[i][j][v] + badd[j];
            if (TANH) x = tanhf(x);
            if (OUTF) Cf[row * ldc + col] = x;
            if (OUTB) Cb[row * ldc + col] = f2bf(x);
          }
        }
      }
    }
}

// ---------------------------------------------------------------------------
// Fused GRU step: acc = [h|x_t] x [Whh|Wih]^T with separate n_x / n_h accumulators.
// Each wave: 32 rows x 16 cols -> two A-frags reuse the 3 gate B-frags
// (6 WMMAs per 32-K step). Block = 4 waves covering 64 output cols.
// grid.z = direction; double-buffered h (hin -> hout) to avoid intra-launch races.
// ---------------------------------------------------------------------------
__global__ __launch_bounds__(128)
void gru_step_kernel(const bf16_t* __restrict__ xseq, int Lseq, int Din,
                     const bf16_t* __restrict__ Wih2,   // [2][1536][Din]
                     const bf16_t* __restrict__ Whh2,   // [2][1536][512]
                     const float* __restrict__ bihF, const float* __restrict__ bhhF,
                     const float* __restrict__ bihB, const float* __restrict__ bhhB,
                     const bf16_t* __restrict__ hin2,   // [2][rowsPad][512]
                     bf16_t* __restrict__ hout2,        // [2][rowsPad][512]
                     bf16_t* __restrict__ hs,           // [rows][Lseq][1024]
                     int rows, int rowsPad, int step) {
  const int lane = threadIdx.x & 31;
  const int wv   = threadIdx.x >> 5;
  const int dir  = blockIdx.z;
  const int t    = dir ? (Lseq - 1 - step) : step;
  const long j0  = ((long)blockIdx.x * 4 + wv) * 16;   // output col tile (0..511)
  const long m0  = (long)blockIdx.y * 32;              // 32 rows per wave
  const int  Hh  = 512;

  const bf16_t* Wih = Wih2 + (long)dir * 1536 * Din;
  const bf16_t* Whh = Whh2 + (long)dir * 1536 * Hh;
  const bf16_t* hin = hin2 + (long)dir * rowsPad * Hh;
  bf16_t*      hout = hout2 + (long)dir * rowsPad * Hh;
  const float* bih = dir ? bihB : bihF;
  const float* bhh = dir ? bhhB : bhhF;

  fx8 accr[2], accz[2], accnh[2], accnx[2];
#pragma unroll
  for (int i = 0; i < 2; ++i) {
    accr[i] = (fx8)0.0f; accz[i] = (fx8)0.0f; accnh[i] = (fx8)0.0f; accnx[i] = (fx8)0.0f;
  }

  // Phase 1: hidden contribution (K = 512)
  for (long k0 = 0; k0 < Hh; k0 += 32) {
    bfx16 a0 = frag_a(hin, m0,      Hh, k0, rows);
    bfx16 a1 = frag_a(hin, m0 + 16, Hh, k0, rows);
    bfx16 br = frag_b(Whh, j0,        Hh, k0, 1536);
    bfx16 bz = frag_b(Whh, j0 + 512,  Hh, k0, 1536);
    bfx16 bn = frag_b(Whh, j0 + 1024, Hh, k0, 1536);
    accr[0]  = wmma_bf16(a0, br, accr[0]);
    accr[1]  = wmma_bf16(a1, br, accr[1]);
    accz[0]  = wmma_bf16(a0, bz, accz[0]);
    accz[1]  = wmma_bf16(a1, bz, accz[1]);
    accnh[0] = wmma_bf16(a0, bn, accnh[0]);
    accnh[1] = wmma_bf16(a1, bn, accnh[1]);
  }
  // Phase 2: input contribution (K = Din), x(m, t, :) with row stride Lseq*Din
  const bf16_t* xrow = xseq + (long)t * Din;
  const long xlda = (long)Lseq * Din;
  for (long k0 = 0; k0 < Din; k0 += 32) {
    bfx16 a0 = frag_a(xrow, m0,      xlda, k0, rows);
    bfx16 a1 = frag_a(xrow, m0 + 16, xlda, k0, rows);
    bfx16 br = frag_b(Wih, j0,        Din, k0, 1536);
    bfx16 bz = frag_b(Wih, j0 + 512,  Din, k0, 1536);
    bfx16 bn = frag_b(Wih, j0 + 1024, Din, k0, 1536);
    accr[0]  = wmma_bf16(a0, br, accr[0]);
    accr[1]  = wmma_bf16(a1, br, accr[1]);
    accz[0]  = wmma_bf16(a0, bz, accz[0]);
    accz[1]  = wmma_bf16(a1, bz, accz[1]);
    accnx[0] = wmma_bf16(a0, bn, accnx[0]);
    accnx[1] = wmma_bf16(a1, bn, accnx[1]);
  }

  const int hi = lane >> 4, nn = lane & 15;
  const long col = j0 + nn;
  const float b_ir = bih[col],        b_hr = bhh[col];
  const float b_iz = bih[col + 512],  b_hz = bhh[col + 512];
  const float b_in = bih[col + 1024], b_hn = bhh[col + 1024];
#pragma unroll
  for (int i = 0; i < 2; ++i)
#pragma unroll
    for (int v = 0; v < 8; ++v) {
      long m = m0 + i * 16 + v + hi * 8;
      if (m < rows) {
        float hp = bf2f(hin[m * Hh + col]);
        float r = sigmoidf_(accr[i][v] + b_ir + b_hr);
        float z = sigmoidf_(accz[i][v] + b_iz + b_hz);
        float n = tanhf(accnx[i][v] + b_in + r * (accnh[i][v] + b_hn));
        float hnew = (1.0f - z) * n + z * hp;
        bf16_t hb = f2bf(hnew);
        hout[m * Hh + col] = hb;
        hs[((long)m * Lseq + t) * 1024 + (long)dir * 512 + col] = hb;
      }
    }
}

// ---------------------------------------------------------------------------
// Attention: scores = q . ctx[t], mask, softmax over T, mix = sum attn[t]*ctx[t]
// mode 0: mask if token==0 (word level);  mode 1: mask if t >= lens[n] (sentence)
// ---------------------------------------------------------------------------
__global__ __launch_bounds__(128)
void attn_mix_kernel(const float* __restrict__ qvec,
                     const bf16_t* __restrict__ ctx, int T,
                     const int* __restrict__ tokens, const int* __restrict__ lens, int mode,
                     bf16_t* __restrict__ mix) {
  const int n = blockIdx.x;
  const int lane = threadIdx.x & 31, wv = threadIdx.x >> 5;
  __shared__ float sc[64];

  for (int t = wv; t < T; t += 4) {
    const bf16_t* c = ctx + ((long)n * T + t) * 1024;
    float s = 0.f;
    for (int k = lane; k < 1024; k += 32) s += qvec[k] * bf2f(c[k]);
#pragma unroll
    for (int off = 16; off > 0; off >>= 1) s += __shfl_xor(s, off, 32);
    bool masked = (mode == 0) ? (tokens[(long)n * T + t] == 0) : (t >= lens[n]);
    if (lane == 0) sc[t] = masked ? -__builtin_inff() : s;
  }
  __syncthreads();
  if (threadIdx.x == 0) {
    float mx = -__builtin_inff();
    for (int t = 0; t < T; ++t) mx = fmaxf(mx, sc[t]);
    float sum = 0.f;
    for (int t = 0; t < T; ++t) { float e = expf(sc[t] - mx); sc[t] = e; sum += e; }
    float inv = 1.f / sum;
    for (int t = 0; t < T; ++t) sc[t] *= inv;
  }
  __syncthreads();
  for (int h = threadIdx.x; h < 1024; h += 128) {
    float acc = 0.f;
    for (int t = 0; t < T; ++t) acc += sc[t] * bf2f(ctx[((long)n * T + t) * 1024 + h]);
    mix[(long)n * 1024 + h] = f2bf(acc);
  }
}

// ---------------------------------------------------------------------------
// starts[b] = prefix sum of doc_lens
// ---------------------------------------------------------------------------
__global__ void starts_kernel(const int* __restrict__ lens, int* __restrict__ starts, int B) {
  if (threadIdx.x == 0) {
    int s = 0;
    for (int b = 0; b < B; ++b) { starts[b] = s; s += lens[b]; }
  }
}

// sent_in[b, p, :] = word_out[starts[b]+p] if p < lens[b] else 0
__global__ void sent_gather_kernel(const bf16_t* __restrict__ wordout,
                                   const int* __restrict__ lens, const int* __restrict__ starts,
                                   bf16_t* __restrict__ sentin, int Lmax) {
  int b = blockIdx.x / Lmax, p = blockIdx.x % Lmax;
  bf16_t* dst = sentin + ((long)b * Lmax + p) * 1024;
  if (p < lens[b]) {
    const bf16_t* src = wordout + (long)(starts[b] + p) * 1024;
    for (int i = threadIdx.x; i < 1024; i += 256) dst[i] = src[i];
  } else {
    bf16_t z = f2bf(0.f);
    for (int i = threadIdx.x; i < 1024; i += 256) dst[i] = z;
  }
}

// pre[b,t] = content.s + s.(Sal@doc) + abs_p[t] + rel_p[b,t] + bias
__global__ __launch_bounds__(128)
void pre_kernel(const bf16_t* __restrict__ sentout, const float* __restrict__ contentv,
                const float* __restrict__ sd, const float* __restrict__ absEmb,
                const float* __restrict__ absW, const float* __restrict__ relEmb,
                const float* __restrict__ relW, const float* __restrict__ biasv,
                const int* __restrict__ lens, float* __restrict__ pre, int Lmax) {
  int idx = blockIdx.x;
  int b = idx / Lmax, t = idx % Lmax;
  int lane = threadIdx.x & 31, wv = threadIdx.x >> 5;
  const bf16_t* so = sentout + (long)idx * 1024;
  const float* sdb = sd + (long)b * 1024;
  float part = 0.f;
  for (int k = threadIdx.x; k < 1024; k += 128) {
    float v = bf2f(so[k]);
    part += v * (contentv[k] + sdb[k]);
  }
#pragma unroll
  for (int off = 16; off > 0; off >>= 1) part += __shfl_xor(part, off, 32);
  __shared__ float red[4];
  if (lane == 0) red[wv] = part;
  __syncthreads();
  if (threadIdx.x == 0) {
    float tot = red[0] + red[1] + red[2] + red[3];
    float ap = 0.f, rp = 0.f;
    for (int i = 0; i < 50; ++i) ap += absEmb[t * 50 + i] * absW[i];
    int len = lens[b];
    float rf = rintf((float)(t + 1) * 9.0f / (float)len);   // round-half-even like jnp.round
    int ri = (int)rf; if (ri < 0) ri = 0; if (ri > 9) ri = 9;
    for (int i = 0; i < 50; ++i) rp += relEmb[ri * 50 + i] * relW[i];
    pre[idx] = tot + ap + rp + biasv[0];
  }
}

// Sequential novelty scan per doc; writes compacted probs at out[starts[b]+t]
__global__ __launch_bounds__(256)
void scan_kernel(const float* __restrict__ pre, const float* __restrict__ hWn,
                 const bf16_t* __restrict__ sentout, const int* __restrict__ lens,
                 const int* __restrict__ starts, float* __restrict__ out, int Lmax) {
  int b = blockIdx.x;
  int lane = threadIdx.x & 31, wv = threadIdx.x >> 5;
  __shared__ float s[1024];
  __shared__ float red[8];
  __shared__ float gate;
  for (int k = threadIdx.x; k < 1024; k += 256) s[k] = 0.f;
  __syncthreads();
  int len = lens[b];
  for (int t = 0; t < len; ++t) {
    const float* hw = hWn + ((long)b * Lmax + t) * 1024;
    float part = 0.f;
    for (int k = threadIdx.x; k < 1024; k += 256) part += hw[k] * tanhf(s[k]);
#pragma unroll
    for (int off = 16; off > 0; off >>= 1) part += __shfl_xor(part, off, 32);
    if (lane == 0) red[wv] = part;
    __syncthreads();
    if (threadIdx.x == 0) {
      float d = 0.f;
      for (int w = 0; w < 8; ++w) d += red[w];
      gate = pre[(long)b * Lmax + t] - d;
    }
    __syncthreads();
    float prob = sigmoidf_(gate);
    const bf16_t* so = sentout + ((long)b * Lmax + t) * 1024;
    for (int k = threadIdx.x; k < 1024; k += 256) s[k] += prob * bf2f(so[k]);
    if (threadIdx.x == 0) out[starts[b] + t] = prob;
    __syncthreads();
  }
}

// ---------------------------------------------------------------------------
// Host orchestration
// ---------------------------------------------------------------------------
extern "C" void kernel_launch(void* const* d_in, const int* in_sizes, int n_in,
                              void* d_out, int out_size, void* d_ws, size_t ws_size,
                              hipStream_t stream) {
  enum { I_EMBED = 0, I_ABS_EMB, I_REL_EMB,
         I_WIH_F, I_WHH_F, I_BIH_F, I_BHH_F,
         I_WIH_B, I_WHH_B, I_BIH_B, I_BHH_B,
         I_SWIH_F, I_SWHH_F, I_SBIH_F, I_SBHH_F,
         I_SWIH_B, I_SWHH_B, I_SBIH_B, I_SBHH_B,
         I_WQ, I_SQ, I_ATTNW, I_ATTNB, I_FCW, I_FCB,
         I_CONTENT, I_SAL, I_NOV, I_ABSW, I_RELW, I_BIAS,
         I_X, I_LENS };

  const int L = 50, Hh = 512, H2 = 1024, D = 256, B = 32, LMAX = 40;
  const int NS  = in_sizes[I_X] / L;          // total sentence count
  const int NSp = (NS + 31) & ~31;            // pad to 32-row GRU tiles

  char* ws = (char*)d_ws;
  size_t off = 0;
  auto alloc = [&](size_t bytes) -> char* {
    char* p = ws + off;
    off = (off + bytes + 255) & ~(size_t)255;
    return p;
  };

  bf16_t* wihW   = (bf16_t*)alloc((size_t)2 * 1536 * D  * 2);
  bf16_t* whhW   = (bf16_t*)alloc((size_t)2 * 1536 * Hh * 2);
  bf16_t* wihS   = (bf16_t*)alloc((size_t)2 * 1536 * H2 * 2);
  bf16_t* whhS   = (bf16_t*)alloc((size_t)2 * 1536 * Hh * 2);
  bf16_t* attnW1 = (bf16_t*)alloc((size_t)H2 * H2 * 2);
  bf16_t* fcWb   = (bf16_t*)alloc((size_t)H2 * H2 * 2);
  bf16_t* salWb  = (bf16_t*)alloc((size_t)H2 * H2 * 2);
  bf16_t* novTb  = (bf16_t*)alloc((size_t)H2 * H2 * 2);
  float*  qbw    = (float*)alloc((size_t)H2 * 4);
  float*  qbs    = (float*)alloc((size_t)H2 * 4);
  bf16_t* xe     = (bf16_t*)alloc((size_t)NSp * L * D * 2);
  bf16_t* hA     = (bf16_t*)alloc((size_t)2 * NSp * Hh * 2);
  bf16_t* hB     = (bf16_t*)alloc((size_t)2 * NSp * Hh * 2);
  bf16_t* woutb  = (bf16_t*)alloc((size_t)NSp * L * H2 * 2);
  bf16_t* mixW   = (bf16_t*)alloc((size_t)NSp * H2 * 2);
  bf16_t* wordob = (bf16_t*)alloc((size_t)NSp * H2 * 2);
  int*    starts = (int*)alloc((size_t)B * 4);
  bf16_t* sentin = (bf16_t*)alloc((size_t)B * LMAX * H2 * 2);
  bf16_t* shA    = (bf16_t*)alloc((size_t)2 * B * Hh * 2);
  bf16_t* shB    = (bf16_t*)alloc((size_t)2 * B * Hh * 2);
  bf16_t* sentob = (bf16_t*)alloc((size_t)B * LMAX * H2 * 2);
  bf16_t* mixS   = (bf16_t*)alloc((size_t)B * H2 * 2);
  bf16_t* docsb  = (bf16_t*)alloc((size_t)B * H2 * 2);
  bf16_t* docvb  = (bf16_t*)alloc((size_t)B * H2 * 2);
  float*  sdF    = (float*)alloc((size_t)B * H2 * 4);
  float*  hWnF   = (float*)alloc((size_t)B * LMAX * H2 * 4);
  float*  preF   = (float*)alloc((size_t)B * LMAX * 4);
  (void)ws_size; (void)n_in; (void)out_size;

  auto convN = [&](const void* src, bf16_t* dst, long rows, long cols, long stride, int tr) {
    long tot = rows * cols;
    conv_bf16_kernel<<<dim3((unsigned)((tot + 255) / 256)), dim3(256), 0, stream>>>(
        (const float*)src, dst, rows, cols, stride, tr);
  };

  // One-time weight conversion to bf16 (fwd/bwd packed along dim 0)
  convN(d_in[I_WIH_F],  wihW,                   1536, D,  D,  0);
  convN(d_in[I_WIH_B],  wihW + (long)1536 * D,  1536, D,  D,  0);
  convN(d_in[I_WHH_F],  whhW,                   1536, Hh, Hh, 0);
  convN(d_in[I_WHH_B],  whhW + (long)1536 * Hh, 1536, Hh, Hh, 0);
  convN(d_in[I_SWIH_F], wihS,                   1536, H2, H2, 0);
  convN(d_in[I_SWIH_B], wihS + (long)1536 * H2, 1536, H2, H2, 0);
  convN(d_in[I_SWHH_F], whhS,                   1536, Hh, Hh, 0);
  convN(d_in[I_SWHH_B], whhS + (long)1536 * Hh, 1536, Hh, Hh, 0);
  convN(d_in[I_ATTNW],  attnW1, H2, H2, 2 * H2, 0);      // first 1024 cols (mix half)
  convN(d_in[I_FCW],    fcWb,   H2, H2, H2, 0);
  convN(d_in[I_SAL],    salWb,  H2, H2, H2, 0);
  convN(d_in[I_NOV],    novTb,  H2, H2, H2, 1);          // transposed: hWn = s @ Nov

  // Constant (query) halves of attn projection folded into bias vectors
  qbias_kernel<<<dim3(8), dim3(128), 0, stream>>>(
      (const float*)d_in[I_ATTNW], (const float*)d_in[I_ATTNB], (const float*)d_in[I_WQ], qbw);
  qbias_kernel<<<dim3(8), dim3(128), 0, stream>>>(
      (const float*)d_in[I_ATTNW], (const float*)d_in[I_ATTNB], (const float*)d_in[I_SQ], qbs);

  // Embedding gather
  embed_gather_kernel<<<dim3((unsigned)(NS * L)), dim3(64), 0, stream>>>(
      (const float*)d_in[I_EMBED], (const int*)d_in[I_X], xe, (long)NS * L);

  // Word-level BiGRU (50 fused steps, both directions per launch)
  hipMemsetAsync(hA, 0, (size_t)2 * NSp * Hh * 2, stream);
  for (int s = 0; s < L; ++s) {
    const bf16_t* hin = (s & 1) ? hB : hA;
    bf16_t*      hout = (s & 1) ? hA : hB;
    gru_step_kernel<<<dim3(Hh / 64, NSp / 32, 2), dim3(128), 0, stream>>>(
        xe, L, D, wihW, whhW,
        (const float*)d_in[I_BIH_F], (const float*)d_in[I_BHH_F],
        (const float*)d_in[I_BIH_B], (const float*)d_in[I_BHH_B],
        hin, hout, woutb, NS, NSp, s);
  }

  // Word attention + output projection (tanh([mix,q]W^T + b))
  attn_mix_kernel<<<dim3((unsigned)NS), dim3(128), 0, stream>>>(
      (const float*)d_in[I_WQ], woutb, L, (const int*)d_in[I_X], nullptr, 0, mixW);
  gemm_bf16_kernel<false, true, true, false, true>
      <<<dim3(H2 / 64, (NS + 127) / 128), dim3(128), 0, stream>>>(
      mixW, H2, attnW1, H2, nullptr, qbw, nullptr, wordob, H2, NS, H2, H2);

  // Sentence packing (zero-pad to LMAX: exact since GRU biases are zero)
  starts_kernel<<<dim3(1), dim3(32), 0, stream>>>((const int*)d_in[I_LENS], starts, B);
  sent_gather_kernel<<<dim3(B * LMAX), dim3(256), 0, stream>>>(
      wordob, (const int*)d_in[I_LENS], starts, sentin, LMAX);

  // Sentence-level BiGRU
  hipMemsetAsync(shA, 0, (size_t)2 * B * Hh * 2, stream);
  for (int s = 0; s < LMAX; ++s) {
    const bf16_t* hin = (s & 1) ? shB : shA;
    bf16_t*      hout = (s & 1) ? shA : shB;
    gru_step_kernel<<<dim3(Hh / 64, B / 32, 2), dim3(128), 0, stream>>>(
        sentin, LMAX, H2, wihS, whhS,
        (const float*)d_in[I_SBIH_F], (const float*)d_in[I_SBHH_F],
        (const float*)d_in[I_SBIH_B], (const float*)d_in[I_SBHH_B],
        hin, hout, sentob, B, B, s);
  }

  // Doc attention (ctx = sent_in per reference) + projections
  attn_mix_kernel<<<dim3(B), dim3(128), 0, stream>>>(
      (const float*)d_in[I_SQ], sentin, LMAX, nullptr, (const int*)d_in[I_LENS], 1, mixS);
  gemm_bf16_kernel<false, true, true, false, true>                         // docs
      <<<dim3(H2 / 64, 1), dim3(128), 0, stream>>>(
      mixS, H2, attnW1, H2, nullptr, qbs, nullptr, docsb, H2, B, H2, H2);
  gemm_bf16_kernel<true, false, true, false, true>                         // doc_vec = tanh(fc)
      <<<dim3(H2 / 64, 1), dim3(128), 0, stream>>>(
      docsb, H2, fcWb, H2, (const float*)d_in[I_FCB], nullptr, nullptr, docvb, H2, B, H2, H2);
  gemm_bf16_kernel<false, false, false, true, false>                       // sd = doc_vec @ Sal^T
      <<<dim3(H2 / 64, 1), dim3(128), 0, stream>>>(
      docvb, H2, salWb, H2, nullptr, nullptr, sdF, nullptr, H2, B, H2, H2);
  gemm_bf16_kernel<false, false, false, true, false>                       // hWn = sent_out @ Nov
      <<<dim3(H2 / 64, (B * LMAX + 127) / 128), dim3(128), 0, stream>>>(
      sentob, H2, novTb, H2, nullptr, nullptr, hWnF, nullptr, H2, B * LMAX, H2, H2);

  // Scalar score terms + sequential novelty scan
  pre_kernel<<<dim3(B * LMAX), dim3(128), 0, stream>>>(
      sentob, (const float*)d_in[I_CONTENT], sdF,
      (const float*)d_in[I_ABS_EMB], (const float*)d_in[I_ABSW],
      (const float*)d_in[I_REL_EMB], (const float*)d_in[I_RELW],
      (const float*)d_in[I_BIAS], (const int*)d_in[I_LENS], preF, LMAX);
  scan_kernel<<<dim3(B), dim3(256), 0, stream>>>(
      preF, hWnF, sentob, (const int*)d_in[I_LENS], starts, (float*)d_out, LMAX);
}
